// ConvblockWithTarget_72919954751798
// MI455X (gfx1250) — matrix operand order repaired and checked by reading, not compiled, not run on hardware
//
#include <hip/hip_runtime.h>
#include <hip/hip_bf16.h>
#include <stddef.h>

// ---- problem constants -----------------------------------------------------
#define BB      4
#define LL      32768
#define CC      64
#define KK      7
#define TT      16378          // (L - 2K)/2 + 1
#define MROWS   (BB*TT)        // 65512
#define EPSV    1e-5f

// ---- WMMA types ------------------------------------------------------------
typedef __attribute__((ext_vector_type(16))) __bf16 v16bf;
typedef __attribute__((ext_vector_type(8)))  float  v8f;

union FragU { uint4 u4[2]; v16bf v; };

static __device__ __forceinline__ unsigned short f2bf(float f) {
    unsigned int u = __float_as_uint(f);
    u = u + 0x7FFFu + ((u >> 16) & 1u);      // round-to-nearest-even
    return (unsigned short)(u >> 16);
}

static __device__ __forceinline__ float fast_tanh(float x) {
    float r;
    // gfx1250 transcendental; v_nop satisfies the 1-op trans hazard gap
    asm volatile("v_tanh_f32 %0, %1\n\tv_nop" : "=v"(r) : "v"(x));
    return r;
}

// ---- ws layout (bytes) -----------------------------------------------------
// [0,2048)        : 512 floats of stats:
//                   sum1@0 sumsq1@64 a1@128 b1@192 sum2@256 sumsq2@320 a2@384 b2@448
// [2048,59392)    : wfragA : 28 nt x 2 kc x 32 lanes x 16 bf16  (weights B-frags)
// [59392,67584)   : wfragC :  4 nt x 2 kc x 32 lanes x 16 bf16  (w_conv  B-frags)
// [67584, +M*64*4): z buffer (pre-BN2 conv output)

// ---------------------------------------------------------------------------
__global__ void k_zero(float* p) {
    int i = blockIdx.x * 256 + threadIdx.x;
    if (i < 512) p[i] = 0.0f;
}

// Pre-pack B-matrix fragments for wave32 WMMA in contiguous k order (same order as the A loads).
// B 32x16 bf16 layout: lane<16 -> col N=lane, K = kc*32 + 0..15 (packed 2/VGPR);
//                      lane>=16 -> col N=lane-16, K = kc*32 + 16..31.
__global__ void k_prep(const float* __restrict__ weights,
                       const float* __restrict__ wconv,
                       unsigned short* __restrict__ wfragA,
                       unsigned short* __restrict__ wfragC) {
    int t = blockIdx.x * 256 + threadIdx.x;
    if (t < 28 * 2 * 32) {                       // weights: 64 x 448 matrix
        int lane = t & 31, kc = (t >> 5) & 1, nt = t >> 6;
        int col = nt * 16 + (lane & 15);         // col = d*7 + k
        int d   = (int)((unsigned)col / 7u);
        int k   = col - d * 7;
        int cb  = kc * 32 + (lane >> 4) * 16;    // K (=c') base, 16 contiguous
        alignas(16) unsigned short tmp[16];
        #pragma unroll
        for (int j = 0; j < 16; ++j)
            tmp[j] = f2bf(weights[((size_t)d * 64 + (cb + j)) * 7 + k]);
        uint4* dst = (uint4*)(wfragA + (size_t)t * 16);
        dst[0] = *(const uint4*)&tmp[0];
        dst[1] = *(const uint4*)&tmp[8];
    }
    if (t < 4 * 2 * 32) {                        // w_conv: 64 x 64
        int lane = t & 31, kc = (t >> 5) & 1, nt = t >> 6;
        int col = nt * 16 + (lane & 15);
        int cb  = kc * 32 + (lane >> 4) * 16;
        alignas(16) unsigned short tmp[16];
        #pragma unroll
        for (int j = 0; j < 16; ++j)
            tmp[j] = f2bf(wconv[(size_t)(cb + j) * 64 + col]);
        uint4* dst = (uint4*)(wfragC + (size_t)t * 16);
        dst[0] = *(const uint4*)&tmp[0];
        dst[1] = *(const uint4*)&tmp[8];
    }
}

// ---------------------------------------------------------------------------
// Pass A: g = x @ W' (WMMA bf16), tanh, gather-multiply, reduce -> y (pre-BN1)
// Block: 64 GEMM rows (l0..l0+63) = 32 output timesteps. 8 waves, 256 threads.
__global__ void __launch_bounds__(256)
k_passA(const float* __restrict__ x,
        const unsigned short* __restrict__ wfragA,
        float* __restrict__ y,
        float* __restrict__ stats) {
    const int chunk = blockIdx.x;                 // 0..511
    const int b     = blockIdx.y;                 // 0..3
    const int l0    = 12 + 64 * chunk;
    const int validRows = (LL - l0 < 64) ? (LL - l0) : 64;
    const bool fullBlk  = (validRows == 64);      // uniform: true for 511/512 chunks
    const int ti0   = 32 * chunk;
    const int validTi = (TT - ti0 < 32) ? (TT - ti0) : 32;

    __shared__ __align__(16) float          xg [76][68];  // f32, rows l0-12 .. l0+63
    __shared__ __align__(16) unsigned short xbf[64][72];  // bf16 A staging rows l0..l0+63
    __shared__ __align__(16) float          yac[32][68];  // y accumulator (residual-seeded)
    __shared__ float ssum[64], ssq[64];

    const int tid = threadIdx.x;
    const float* xb = x + (size_t)b * LL * 64;

    for (int idx = tid; idx < 76 * 64; idx += 256) {
        int r = idx >> 6, c = idx & 63;
        int l = l0 - 12 + r; if (l > LL - 1) l = LL - 1;
        xg[r][c] = xb[(size_t)l * 64 + c];
    }
    __syncthreads();
    for (int idx = tid; idx < 64 * 64; idx += 256) {
        int r = idx >> 6, c = idx & 63;
        xbf[r][c] = f2bf(xg[r + 12][c]);
    }
    for (int idx = tid; idx < 32 * 64; idx += 256) {
        int r = idx >> 6, c = idx & 63;
        yac[r][c] = xg[2 * r + 12][c];            // residual x[:, ::2][-t:]
    }
    if (tid < 64) { ssum[tid] = 0.0f; ssq[tid] = 0.0f; }
    __syncthreads();

    const int wave = tid >> 5, lane = tid & 31;
    const int half = lane >> 4, ln = lane & 15;

    #pragma unroll 2
    for (int t8 = 0; t8 < 14; ++t8) {
        const int tile = wave * 14 + t8;          // 112 C-tiles: 4 M x 28 N
        const int mt = tile / 28, nt = tile % 28;
        v8f acc = {0.f,0.f,0.f,0.f,0.f,0.f,0.f,0.f};
        #pragma unroll
        for (int kc = 0; kc < 2; ++kc) {
            FragU A, Bv;
            // A 16x32 bf16: lane<16 -> row=ln, K 0-15; lane>=16 -> K 16-31
            const unsigned short* ap = &xbf[mt * 16 + ln][kc * 32 + half * 16];
            A.u4[0] = *(const uint4*)ap;
            A.u4[1] = *(const uint4*)(ap + 8);
            const uint4* bp = (const uint4*)(wfragA + (size_t)(((nt * 2 + kc) * 32 + lane)) * 16);
            Bv.u4[0] = bp[0]; Bv.u4[1] = bp[1];
            acc = __builtin_amdgcn_wmma_f32_16x16x32_bf16(
                      false, A.v, false, Bv.v, (short)0, acc, false, false);
        }
        const int col = nt * 16 + ln;             // = d*7 + k
        const int d   = (int)((unsigned)col / 7u);
        const int k   = col - d * 7;
        const int rlB = mt * 16 + half * 8;       // local GEMM row base (even)

        float th[8], xv[8];
        #pragma unroll
        for (int i = 0; i < 8; ++i) th[i] = fast_tanh(acc[i]);
        #pragma unroll
        for (int i = 0; i < 8; ++i) xv[i] = xg[rlB + i + 2 * k][d];

        if (fullBlk) {
            // fast path: no guards; pair rows sharing the same output timestep
            #pragma unroll
            for (int p = 0; p < 4; ++p) {
                float cs = th[2*p] * xv[2*p] + th[2*p+1] * xv[2*p+1];
                atomicAdd(&yac[(rlB >> 1) + p][d], cs);   // ds_add_f32
            }
        } else {
            #pragma unroll
            for (int p = 0; p < 4; ++p) {
                const int rl = rlB + 2 * p;
                float cs = 0.0f;
                if (rl     < validRows) cs += th[2*p]   * xv[2*p];
                if (rl + 1 < validRows) cs += th[2*p+1] * xv[2*p+1];
                if (rl < validRows) atomicAdd(&yac[rl >> 1][d], cs);
            }
        }
    }
    __syncthreads();

    float s = 0.0f, q = 0.0f;
    for (int idx = tid; idx < 32 * 64; idx += 256) {   // c stays fixed per thread
        int r = idx >> 6, c = idx & 63;
        if (r < validTi) {
            float v = yac[r][c];
            y[((size_t)b * TT + ti0 + r) * 64 + c] = v;
            s += v; q += v * v;
        }
    }
    atomicAdd(&ssum[tid & 63], s);
    atomicAdd(&ssq [tid & 63], q);
    __syncthreads();
    if (tid < 64) {
        atomicAdd(&stats[tid],      ssum[tid]);   // sum1
        atomicAdd(&stats[64 + tid], ssq [tid]);   // sumsq1
    }
}

// ---------------------------------------------------------------------------
// Finalize batchnorm params: a = scale*rsqrt(var+eps), b = bias - mean*a
__global__ void k_fin(float* stats, int off,
                      const float* __restrict__ scale,
                      const float* __restrict__ bias, float n) {
    int c = threadIdx.x;
    if (c < 64) {
        float mean = stats[off + c] / n;
        float var  = stats[off + 64 + c] / n - mean * mean;
        float a    = scale[c] * rsqrtf(var + EPSV);
        stats[off + 128 + c] = a;
        stats[off + 192 + c] = bias[c] - mean * a;
    }
}

// ---------------------------------------------------------------------------
// Pass C: y_n = BN1(y); z = y_n @ w_conv via WMMA; BN2 stats.
// Block: 128 rows, 8 waves (wave = M-tile), 4 N-tiles each.
__global__ void __launch_bounds__(256)
k_passC(const float* __restrict__ y,
        const unsigned short* __restrict__ wfragC,
        const float* __restrict__ stats,
        float* __restrict__ z,
        float* __restrict__ sums) {
    const int R0 = blockIdx.x * 128;
    const bool fullBlk = (R0 + 128 <= MROWS);     // uniform: true for 511/512 blocks
    __shared__ __align__(16) unsigned short ybf[128][72];
    __shared__ float a1s[64], b1s[64], zs[64], zq[64];
    const int tid = threadIdx.x;
    if (tid < 64) {
        a1s[tid] = stats[128 + tid]; b1s[tid] = stats[192 + tid];
        zs[tid] = 0.0f; zq[tid] = 0.0f;
    }
    __syncthreads();
    for (int idx = tid; idx < 128 * 64; idx += 256) {
        int r = idx >> 6, c = idx & 63;
        int row = R0 + r; if (row > MROWS - 1) row = MROWS - 1;
        ybf[r][c] = f2bf(y[(size_t)row * 64 + c] * a1s[c] + b1s[c]);
    }
    __syncthreads();

    const int wave = tid >> 5, lane = tid & 31;
    const int half = lane >> 4, ln = lane & 15;
    const int mt = wave;
    #pragma unroll
    for (int nt = 0; nt < 4; ++nt) {
        v8f acc = {0.f,0.f,0.f,0.f,0.f,0.f,0.f,0.f};
        #pragma unroll
        for (int kc = 0; kc < 2; ++kc) {
            FragU A, Bv;
            const unsigned short* ap = &ybf[mt * 16 + ln][kc * 32 + half * 16];
            A.u4[0] = *(const uint4*)ap;
            A.u4[1] = *(const uint4*)(ap + 8);
            const uint4* bp = (const uint4*)(wfragC + (size_t)(((nt * 2 + kc) * 32 + lane)) * 16);
            Bv.u4[0] = bp[0]; Bv.u4[1] = bp[1];
            acc = __builtin_amdgcn_wmma_f32_16x16x32_bf16(
                      false, A.v, false, Bv.v, (short)0, acc, false, false);
        }
        const int col = nt * 16 + ln;
        const size_t rbase = (size_t)(R0 + mt * 16 + half * 8);
        float s = 0.0f, q = 0.0f;
        if (fullBlk) {
            #pragma unroll
            for (int i = 0; i < 8; ++i) {
                float v = acc[i];
                z[(rbase + i) * 64 + col] = v;
                s += v; q += v * v;
            }
        } else {
            #pragma unroll
            for (int i = 0; i < 8; ++i) {
                if (rbase + i < (size_t)MROWS) {
                    float v = acc[i];
                    z[(rbase + i) * 64 + col] = v;
                    s += v; q += v * v;
                }
            }
        }
        atomicAdd(&zs[col], s);
        atomicAdd(&zq[col], q);
    }
    __syncthreads();
    if (tid < 64) {
        atomicAdd(&sums[256 + tid], zs[tid]);     // sum2
        atomicAdd(&sums[320 + tid], zq[tid]);     // sumsq2
    }
}

// ---------------------------------------------------------------------------
// Pass E: out = BN1(y) + gelu_tanh(BN2(z))
__global__ void k_passE(float* __restrict__ out,
                        const float* __restrict__ z,
                        const float* __restrict__ stats) {
    int idx = blockIdx.x * 256 + threadIdx.x;
    if (idx < MROWS * 64) {
        int c = idx & 63;
        float yn = out[idx] * stats[128 + c] + stats[192 + c];
        float zn = z[idx]   * stats[384 + c] + stats[448 + c];
        float u  = 0.7978845608028654f * (zn + 0.044715f * zn * zn * zn);
        float g  = 0.5f * zn * (1.0f + fast_tanh(u));
        out[idx] = yn + g;
    }
}

// ---------------------------------------------------------------------------
extern "C" void kernel_launch(void* const* d_in, const int* in_sizes, int n_in,
                              void* d_out, int out_size, void* d_ws, size_t ws_size,
                              hipStream_t stream) {
    const float* x       = (const float*)d_in[0];
    const float* weights = (const float*)d_in[1];
    const float* w_conv  = (const float*)d_in[2];
    const float* scale1  = (const float*)d_in[3];
    const float* bias1   = (const float*)d_in[4];
    const float* scale2  = (const float*)d_in[5];
    const float* bias2   = (const float*)d_in[6];
    float* out = (float*)d_out;

    float*          stats  = (float*)d_ws;
    unsigned short* wfragA = (unsigned short*)((char*)d_ws + 2048);
    unsigned short* wfragC = (unsigned short*)((char*)d_ws + 59392);
    float*          z      = (float*)((char*)d_ws + 67584);

    k_zero<<<2, 256, 0, stream>>>(stats);
    k_prep<<<7, 256, 0, stream>>>(weights, w_conv, wfragA, wfragC);

    dim3 gA(512, 4);
    k_passA<<<gA, 256, 0, stream>>>(x, wfragA, out, stats);
    k_fin<<<1, 64, 0, stream>>>(stats, 0, scale1, bias1, (float)MROWS);
    k_passC<<<512, 256, 0, stream>>>(out, wfragC, stats, z, stats);
    k_fin<<<1, 64, 0, stream>>>(stats, 256, scale2, bias2, (float)MROWS);

    int n = MROWS * 64;
    k_passE<<<(n + 255) / 256, 256, 0, stream>>>(out, z, stats);
}
